// TransformerLayer_23484881175084
// MI455X (gfx1250) — compile-verified, matrix operand order checked by reading, not run on hardware
//
#include <hip/hip_runtime.h>
#include <math.h>

// ---------------- Problem constants ----------------
constexpr int Bc  = 2;
constexpr int Nc  = 2048;
constexpr int Dc  = 256;
constexpr int Hc  = 4;
constexpr int HDc = 64;
constexpr int BHc = Bc * Hc;      // 8
constexpr int BNc = Bc * Nc;      // 4096

// ---------------- WMMA types / helpers ----------------
typedef __attribute__((ext_vector_type(16))) _Float16 v16h;
typedef __attribute__((ext_vector_type(8)))  float    v8f;

union F16x16 {
    v16h     v;
    unsigned u[8];
    _Float16 h[16];
};

// Pack two f32 -> adjacent f16 in one dword.
__device__ __forceinline__ unsigned pack2(float x, float y) {
    union { unsigned u; _Float16 h[2]; } p;
    p.h[0] = (_Float16)x;
    p.h[1] = (_Float16)y;
    return p.u;
}

// K index of first element held in 32-bit slot v for a 16-bit A/B fragment.
// Lanes 0-15: K halves {0..7,16..23}; lanes 16-31: {8..15,24..31}.
__device__ __forceinline__ int KB(int v, int khalf) {
    return (v < 4 ? 2 * v : 16 + 2 * (v - 4)) + 8 * khalf;
}

__device__ __forceinline__ v8f wmma_f16(const v16h& a, const v16h& b, const v8f& c) {
    return __builtin_amdgcn_wmma_f32_16x16x32_f16(false, a, false, b, (short)0, c, false, false);
}

// =====================================================================
// Generic WMMA GEMM:  out[M,Nn] = A[M,K] @ Bw[K,Nn] + bias (+ resid)
// Block = 128 threads (4 waves), tile 64x64, LDS-staged f32->f16 with
// packed 32-bit DS stores and 64-bit coalesced global loads for A.
// =====================================================================
__global__ __launch_bounds__(128)
void wmma_gemm_kernel(const float* __restrict__ A, const float* __restrict__ Bw,
                      const float* __restrict__ bias, const float* __restrict__ resid,
                      float* __restrict__ out,
                      int M, int K, int Nn, int ldOut)
{
    __shared__ _Float16 a_s[64 * 32];   // [row][k]
    __shared__ _Float16 b_s[64 * 32];   // [col][k]  (transposed for pair loads)

    const int tid   = threadIdx.x;
    const int wave  = tid >> 5;
    const int lane  = tid & 31;
    const int m0    = blockIdx.x * 64;
    const int n0    = blockIdx.y * 64;
    const int mfrag = lane & 15;
    const int khalf = lane >> 4;

    const v8f z8 = {0.f, 0.f, 0.f, 0.f, 0.f, 0.f, 0.f, 0.f};
    v8f acc[4];
    acc[0] = z8; acc[1] = z8; acc[2] = z8; acc[3] = z8;

    for (int k0 = 0; k0 < K; k0 += 32) {
        // ---- stage A tile (64 rows x 32 k): float2 loads, packed b32 DS stores ----
        {
            int kp = tid & 15;          // k-pair 0..15  (kk = 2*kp)
            int r  = tid >> 4;          // 0..7
            if (k0 + 32 < K) __builtin_prefetch(&A[(size_t)(m0 + r) * K + k0 + 32], 0, 1);
            #pragma unroll
            for (int it = 0; it < 8; ++it, r += 8) {
                const float2 p = *(const float2*)&A[(size_t)(m0 + r) * K + k0 + 2 * kp];
                *(unsigned*)&a_s[r * 32 + 2 * kp] = pack2(p.x, p.y);
            }
        }
        // ---- stage B tile transposed: b_s[col][k]; coalesced loads along col,
        //      two k-rows packed into one b32 DS store ----
        {
            int c  = tid & 63;          // 0..63
            int kp = tid >> 6;          // 0..1 (kk = 2*kp)
            #pragma unroll
            for (int it = 0; it < 8; ++it, kp += 2) {
                float b0 = Bw[(size_t)(k0 + 2 * kp) * Nn + (n0 + c)];
                float b1 = Bw[(size_t)(k0 + 2 * kp + 1) * Nn + (n0 + c)];
                *(unsigned*)&b_s[c * 32 + 2 * kp] = pack2(b0, b1);
            }
        }
        __syncthreads();

        // ---- build A fragment for this wave's 16 rows ----
        F16x16 af;
        #pragma unroll
        for (int v = 0; v < 8; ++v) {
            int kk = KB(v, khalf);
            af.u[v] = *(const unsigned*)&a_s[(wave * 16 + mfrag) * 32 + kk];
        }
        // ---- 4 x (B fragment + WMMA) covering 64 output columns ----
        #pragma unroll
        for (int nt = 0; nt < 4; ++nt) {
            F16x16 bf;
            #pragma unroll
            for (int v = 0; v < 8; ++v) {
                int kk = KB(v, khalf);
                bf.u[v] = *(const unsigned*)&b_s[(nt * 16 + mfrag) * 32 + kk];
            }
            acc[nt] = wmma_f16(af.v, bf.v, acc[nt]);
        }
        __syncthreads();
    }

    // ---- epilogue: bias (+ residual), store ----
    #pragma unroll
    for (int nt = 0; nt < 4; ++nt) {
        int n = n0 + nt * 16 + mfrag;
        float bv = bias ? bias[n] : 0.f;
        #pragma unroll
        for (int r = 0; r < 8; ++r) {
            int m = m0 + wave * 16 + r + 8 * khalf;
            float vout = acc[nt][r] + bv;
            if (resid) vout += resid[(size_t)m * Nn + n];
            out[(size_t)m * ldOut + n] = vout;
        }
    }
}

// =====================================================================
// Attention scores: sim[bh,i,j] = scale * Q[bh,i,:] . K[bh,j,:]   (HD=64)
// 1 wave computes a 16x256 strip (16 j-tiles), 2 chained WMMAs each.
// grid = (N/16, N/256, B*H), block = 32
// =====================================================================
__global__ __launch_bounds__(32)
void attn_scores_kernel(const _Float16* __restrict__ Qh, const _Float16* __restrict__ Kh,
                        float* __restrict__ simOut, float scale)
{
    const int lane  = threadIdx.x;
    const int i0    = blockIdx.x * 16;
    const int jg    = blockIdx.y;
    const int bh    = blockIdx.z;
    const int mfrag = lane & 15;
    const int khalf = lane >> 4;
    const unsigned* Qu = (const unsigned*)Qh;
    const unsigned* Ku = (const unsigned*)Kh;
    const v8f z8 = {0.f, 0.f, 0.f, 0.f, 0.f, 0.f, 0.f, 0.f};

    F16x16 a0, a1;
    #pragma unroll
    for (int v = 0; v < 8; ++v) {
        int kk = KB(v, khalf);
        size_t base = ((size_t)bh * Nc + i0 + mfrag) * HDc;
        a0.u[v] = Qu[(base + kk) >> 1];
        a1.u[v] = Qu[(base + kk + 32) >> 1];
    }

    for (int jt = 0; jt < 16; ++jt) {
        int j0 = jg * 256 + jt * 16;
        F16x16 b0, b1;
        #pragma unroll
        for (int v = 0; v < 8; ++v) {
            int kk = KB(v, khalf);
            size_t base = ((size_t)bh * Nc + j0 + mfrag) * HDc;
            b0.u[v] = Ku[(base + kk) >> 1];
            b1.u[v] = Ku[(base + kk + 32) >> 1];
        }
        v8f c = z8;
        c = wmma_f16(a0.v, b0.v, c);
        c = wmma_f16(a1.v, b1.v, c);
        #pragma unroll
        for (int r = 0; r < 8; ++r) {
            int m = i0 + r + 8 * khalf;
            simOut[((size_t)bh * Nc + m) * Nc + j0 + mfrag] = c[r] * scale;
        }
    }
}

// =====================================================================
// ctx = attn @ V, written merged as [B,N,D] (col = h*HD+d).
// Vt layout: [B,H,HD,N] f16 so K-pairs are contiguous.
// grid = (N/16, B*H), block = 32
// =====================================================================
__global__ __launch_bounds__(32)
void attn_av_kernel(const float* __restrict__ attn, const _Float16* __restrict__ Vt,
                    float* __restrict__ outMerged)
{
    const int lane  = threadIdx.x;
    const int i0    = blockIdx.x * 16;
    const int bh    = blockIdx.y;
    const int b     = bh / Hc;
    const int h     = bh % Hc;
    const int mfrag = lane & 15;
    const int khalf = lane >> 4;
    const unsigned* Vu = (const unsigned*)Vt;
    const v8f z8 = {0.f, 0.f, 0.f, 0.f, 0.f, 0.f, 0.f, 0.f};
    v8f acc[4];
    acc[0] = z8; acc[1] = z8; acc[2] = z8; acc[3] = z8;

    for (int k0 = 0; k0 < Nc; k0 += 32) {
        F16x16 af;
        #pragma unroll
        for (int v = 0; v < 8; ++v) {
            int kk = KB(v, khalf);
            const float2 p =
                *(const float2*)&attn[((size_t)bh * Nc + i0 + mfrag) * Nc + k0 + kk];
            af.u[v] = pack2(p.x, p.y);
        }
        #pragma unroll
        for (int nt = 0; nt < 4; ++nt) {
            F16x16 bf;
            #pragma unroll
            for (int v = 0; v < 8; ++v) {
                int kk = KB(v, khalf);
                int c  = nt * 16 + mfrag;
                bf.u[v] = Vu[(((size_t)bh * HDc + c) * Nc + k0 + kk) >> 1];
            }
            acc[nt] = wmma_f16(af.v, bf.v, acc[nt]);
        }
    }
    #pragma unroll
    for (int nt = 0; nt < 4; ++nt) {
        int c = nt * 16 + mfrag;
        #pragma unroll
        for (int r = 0; r < 8; ++r) {
            int i = i0 + r + 8 * khalf;
            outMerged[((size_t)b * Nc + i) * Dc + h * HDc + c] = acc[nt][r];
        }
    }
}

// =====================================================================
// Rotary + head split from qkv buffer [B,N,3D] (channel = (h*HD+d)*3+s).
// Writes Qh,Kh [B,H,N,HD] f16 (packed dword stores) and Vt [B,H,HD,N] f16.
// =====================================================================
__global__ void rotary_split_kernel(const float* __restrict__ qkv, const float* __restrict__ enc,
                                    _Float16* __restrict__ Qh, _Float16* __restrict__ Kh,
                                    _Float16* __restrict__ Vt)
{
    int idx = blockIdx.x * blockDim.x + threadIdx.x;   // over B*H*N*(HD/2)
    if (idx >= BHc * Nc * (HDc / 2)) return;
    int p  = idx % (HDc / 2);
    int t  = idx / (HDc / 2);
    int n  = t % Nc;
    int bh = t / Nc;
    int b  = bh / Hc, h = bh % Hc;
    int d0 = 2 * p, d1 = d0 + 1;

    const float* row = qkv + (size_t)(b * Nc + n) * (3 * Dc);
    float q0 = row[(h * HDc + d0) * 3 + 0], q1 = row[(h * HDc + d1) * 3 + 0];
    float k0 = row[(h * HDc + d0) * 3 + 1], k1 = row[(h * HDc + d1) * 3 + 1];
    float v0 = row[(h * HDc + d0) * 3 + 2], v1 = row[(h * HDc + d1) * 3 + 2];

    size_t ebase = ((size_t)bh * Nc + n) * HDc;
    const float* e0 = enc + ebase;                               // freqs[0]
    const float* e1 = enc + (size_t)BHc * Nc * HDc + ebase;      // freqs[1]
    float f00 = e0[d0], f01 = e0[d1], f10 = e1[d0], f11 = e1[d1];

    // out[2i] = x[2i]*f0[2i] - x[2i+1]*f1[2i];  out[2i+1] = x[2i+1]*f0[2i+1] + x[2i]*f1[2i+1]
    float qr0 = q0 * f00 - q1 * f10;
    float qr1 = q1 * f01 + q0 * f11;
    float kr0 = k0 * f00 - k1 * f10;
    float kr1 = k1 * f01 + k0 * f11;

    size_t obase = ((size_t)bh * Nc + n) * HDc;
    *(unsigned*)&Qh[obase + d0] = pack2(qr0, qr1);
    *(unsigned*)&Kh[obase + d0] = pack2(kr0, kr1);
    Vt[((size_t)bh * HDc + d0) * Nc + n] = (_Float16)v0;
    Vt[((size_t)bh * HDc + d1) * Nc + n] = (_Float16)v1;
}

// =====================================================================
// Head split [B,N,D] -> f16 [B,H,N,HD] (dstNH) and/or [B,H,HD,N] (dstT)
// One thread per d-pair so dstNH stores are packed dwords.
// =====================================================================
__global__ void split_heads_kernel(const float* __restrict__ src,
                                   _Float16* __restrict__ dstNH,
                                   _Float16* __restrict__ dstT, float scale)
{
    int idx = blockIdx.x * blockDim.x + threadIdx.x;   // over B*N*D/2
    if (idx >= BNc * Dc / 2) return;
    int dp = idx % (Dc / 2);
    int n  = (idx / (Dc / 2)) % Nc;
    int b  = idx / ((Dc / 2) * Nc);
    int d0 = 2 * dp;
    int h  = d0 / HDc, c = d0 % HDc;
    int bh = b * Hc + h;
    const float2 p = *(const float2*)&src[((size_t)(b * Nc + n)) * Dc + d0];
    float v0 = p.x * scale, v1 = p.y * scale;
    if (dstNH) *(unsigned*)&dstNH[((size_t)bh * Nc + n) * HDc + c] = pack2(v0, v1);
    if (dstT) {
        dstT[((size_t)bh * HDc + c) * Nc + n]     = (_Float16)v0;
        dstT[((size_t)bh * HDc + c + 1) * Nc + n] = (_Float16)v1;
    }
}

// =====================================================================
// Row softmax in place over last dim (N). grid = B*H*N, block = 256.
// =====================================================================
__global__ __launch_bounds__(256)
void row_softmax_kernel(float* __restrict__ attn)
{
    __shared__ float red[256];
    float* p = attn + (size_t)blockIdx.x * Nc;
    int t = threadIdx.x;

    float v[8];
    float mx = -3.4e38f;
    #pragma unroll
    for (int s = 0; s < 8; ++s) { v[s] = p[t + 256 * s]; mx = fmaxf(mx, v[s]); }
    red[t] = mx; __syncthreads();
    for (int o = 128; o > 0; o >>= 1) { if (t < o) red[t] = fmaxf(red[t], red[t + o]); __syncthreads(); }
    mx = red[0]; __syncthreads();

    float sum = 0.f;
    #pragma unroll
    for (int s = 0; s < 8; ++s) { v[s] = expf(v[s] - mx); sum += v[s]; }
    red[t] = sum; __syncthreads();
    for (int o = 128; o > 0; o >>= 1) { if (t < o) red[t] += red[t + o]; __syncthreads(); }
    float inv = 1.f / red[0];

    #pragma unroll
    for (int s = 0; s < 8; ++s) p[t + 256 * s] = v[s] * inv;
}

// =====================================================================
// Column softmax + transpose: attnT[bh,j,i] = softmax_i(sim[bh,i,j]).
// grid = B*H*N (one column each), block = 256.
// =====================================================================
__global__ __launch_bounds__(256)
void col_softmax_t_kernel(const float* __restrict__ sim, float* __restrict__ attnT)
{
    __shared__ float col[Nc];
    __shared__ float red[256];
    int j  = blockIdx.x % Nc;
    int bh = blockIdx.x / Nc;
    const float* s = sim + (size_t)bh * Nc * Nc;
    int t = threadIdx.x;

    float mx = -3.4e38f;
    for (int i = t; i < Nc; i += 256) { float x = s[(size_t)i * Nc + j]; col[i] = x; mx = fmaxf(mx, x); }
    red[t] = mx; __syncthreads();
    for (int o = 128; o > 0; o >>= 1) { if (t < o) red[t] = fmaxf(red[t], red[t + o]); __syncthreads(); }
    mx = red[0]; __syncthreads();

    float sum = 0.f;
    for (int i = t; i < Nc; i += 256) { float e = expf(col[i] - mx); col[i] = e; sum += e; }
    red[t] = sum; __syncthreads();
    for (int o = 128; o > 0; o >>= 1) { if (t < o) red[t] += red[t + o]; __syncthreads(); }
    float inv = 1.f / red[0];
    __syncthreads();

    float* o = attnT + ((size_t)bh * Nc + j) * Nc;
    for (int i = t; i < Nc; i += 256) o[i] = col[i] * inv;
}

// =====================================================================
// LayerNorm (over 2D=512) + exact GELU, in place. grid = B*N, block = 256.
// =====================================================================
__global__ __launch_bounds__(256)
void ln_gelu_kernel(float* __restrict__ hbuf, const float* __restrict__ g,
                    const float* __restrict__ bn)
{
    __shared__ float red[256];
    float* p = hbuf + (size_t)blockIdx.x * (2 * Dc);
    int t = threadIdx.x;
    float x0 = p[t], x1 = p[t + 256];

    red[t] = x0 + x1; __syncthreads();
    for (int o = 128; o > 0; o >>= 1) { if (t < o) red[t] += red[t + o]; __syncthreads(); }
    float mean = red[0] * (1.f / 512.f); __syncthreads();

    float d0 = x0 - mean, d1 = x1 - mean;
    red[t] = d0 * d0 + d1 * d1; __syncthreads();
    for (int o = 128; o > 0; o >>= 1) { if (t < o) red[t] += red[t + o]; __syncthreads(); }
    float inv = rsqrtf(red[0] * (1.f / 512.f) + 1e-5f);

    float y0 = d0 * inv * g[t] + bn[t];
    float y1 = d1 * inv * g[t + 256] + bn[t + 256];
    p[t]       = 0.5f * y0 * (1.f + erff(y0 * 0.70710678118f));
    p[t + 256] = 0.5f * y1 * (1.f + erff(y1 * 0.70710678118f));
}

// =====================================================================
// Copy x [B,N,D] into first half of concat buffer [B,N,2D].
// =====================================================================
__global__ void copy_to_cat_kernel(const float* __restrict__ x, float* __restrict__ cat)
{
    int idx = blockIdx.x * blockDim.x + threadIdx.x;
    if (idx >= BNc * Dc) return;
    int m = idx / Dc, c = idx % Dc;
    cat[(size_t)m * (2 * Dc) + c] = x[idx];
}

// =====================================================================
// Host-side orchestration
// =====================================================================
extern "C" void kernel_launch(void* const* d_in, const int* in_sizes, int n_in,
                              void* d_out, int out_size, void* d_ws, size_t ws_size,
                              hipStream_t stream)
{
    (void)in_sizes; (void)n_in; (void)out_size; (void)ws_size;

    const float* desc0 = (const float*)d_in[0];
    const float* desc1 = (const float*)d_in[1];
    const float* enc0  = (const float*)d_in[2];
    const float* enc1  = (const float*)d_in[3];
    const float* sa_wqkv = (const float*)d_in[4];
    const float* sa_bqkv = (const float*)d_in[5];
    const float* sa_wo = (const float*)d_in[6];
    const float* sa_bo = (const float*)d_in[7];
    const float* sa_w1 = (const float*)d_in[8];
    const float* sa_b1 = (const float*)d_in[9];
    const float* sa_g  = (const float*)d_in[10];
    const float* sa_bn = (const float*)d_in[11];
    const float* sa_w2 = (const float*)d_in[12];
    const float* sa_b2 = (const float*)d_in[13];
    const float* ca_wqk = (const float*)d_in[14];
    const float* ca_bqk = (const float*)d_in[15];
    const float* ca_wv = (const float*)d_in[16];
    const float* ca_bv = (const float*)d_in[17];
    const float* ca_wo = (const float*)d_in[18];
    const float* ca_bo = (const float*)d_in[19];
    const float* ca_w1 = (const float*)d_in[20];
    const float* ca_b1 = (const float*)d_in[21];
    const float* ca_g  = (const float*)d_in[22];
    const float* ca_bn = (const float*)d_in[23];
    const float* ca_w2 = (const float*)d_in[24];
    const float* ca_b2 = (const float*)d_in[25];

    // ---- output regions (return order: x0, x1, attn01, attn10, attn0, attn1) ----
    float* x0_out = (float*)d_out;
    float* x1_out = x0_out + (size_t)BNc * Dc;
    float* attn01 = x1_out + (size_t)BNc * Dc;
    float* attn10 = attn01 + (size_t)BHc * Nc * Nc;
    float* attn0  = attn10 + (size_t)BHc * Nc * Nc;
    float* attn1  = attn0  + (size_t)BHc * Nc * Nc;

    // ---- workspace bump allocator ----
    char* ws = (char*)d_ws;
    size_t off = 0;
    auto bump = [&](size_t bytes) -> void* {
        void* p = ws + off;
        off += (bytes + 255) & ~(size_t)255;
        return p;
    };
    float* W0 = (float*)bump((size_t)BNc * 3 * Dc * sizeof(float));  // qkv / ffn hidden
    float* W1 = (float*)bump((size_t)BNc * 2 * Dc * sizeof(float));  // concat buffer
    float* W2 = (float*)bump((size_t)2 * BNc * Dc * sizeof(float));  // ctx / m0,m1 merged
    const size_t f16sz = (size_t)BHc * Nc * HDc * sizeof(_Float16);
    _Float16* F0 = (_Float16*)bump(f16sz);   // Qh  / qk0h
    _Float16* F1 = (_Float16*)bump(f16sz);   // Kh  / qk1h
    _Float16* F2 = (_Float16*)bump(f16sz);   // Vt  / v0t
    _Float16* F3 = (_Float16*)bump(f16sz);   //       v1t
    float* d0b = (float*)bump((size_t)BNc * Dc * sizeof(float));
    float* d1b = (float*)bump((size_t)BNc * Dc * sizeof(float));
    float* P   = (float*)bump((size_t)BNc * Dc * sizeof(float));     // cross proj temp

    auto gemm = [&](const float* A, const float* Bw, const float* bias, const float* resid,
                    float* out, int M, int K, int Nn, int ldOut) {
        wmma_gemm_kernel<<<dim3(M / 64, Nn / 64), 128, 0, stream>>>(
            A, Bw, bias, resid, out, M, K, Nn, ldOut);
    };

    const float inv_sqrt_hd = 0.125f;            // 64^-0.5
    const float s_cross     = 0.3535533906f;     // 64^-0.25 (applied to both qk0, qk1)

    // ----------------- Self block (shared weights, run per descriptor) -----------------
    auto self_block = [&](const float* x, const float* enc, float* attn, float* dblk) {
        gemm(x, sa_wqkv, sa_bqkv, nullptr, W0, BNc, Dc, 3 * Dc, 3 * Dc);
        rotary_split_kernel<<<(BHc * Nc * (HDc / 2)) / 256, 256, 0, stream>>>(W0, enc, F0, F1, F2);
        attn_scores_kernel<<<dim3(Nc / 16, Nc / 256, BHc), 32, 0, stream>>>(F0, F1, attn, inv_sqrt_hd);
        row_softmax_kernel<<<BHc * Nc, 256, 0, stream>>>(attn);
        attn_av_kernel<<<dim3(Nc / 16, BHc), 32, 0, stream>>>(attn, F2, W2);
        gemm(W2, sa_wo, sa_bo, nullptr, W1 + Dc, BNc, Dc, Dc, 2 * Dc);      // msg -> cat[:,D:]
        copy_to_cat_kernel<<<(BNc * Dc) / 256, 256, 0, stream>>>(x, W1);    // x   -> cat[:,:D]
        gemm(W1, sa_w1, sa_b1, nullptr, W0, BNc, 2 * Dc, 2 * Dc, 2 * Dc);
        ln_gelu_kernel<<<BNc, 256, 0, stream>>>(W0, sa_g, sa_bn);
        gemm(W0, sa_w2, sa_b2, /*resid=*/x, dblk, BNc, 2 * Dc, Dc, Dc);
    };

    self_block(desc0, enc0, attn0, d0b);
    self_block(desc1, enc1, attn1, d1b);

    // ----------------- Cross block -----------------
    const int splitBlocks = (BNc * Dc / 2) / 256;
    gemm(d0b, ca_wqk, ca_bqk, nullptr, P, BNc, Dc, Dc, Dc);
    split_heads_kernel<<<splitBlocks, 256, 0, stream>>>(P, F0, nullptr, s_cross);
    gemm(d1b, ca_wqk, ca_bqk, nullptr, P, BNc, Dc, Dc, Dc);
    split_heads_kernel<<<splitBlocks, 256, 0, stream>>>(P, F1, nullptr, s_cross);
    gemm(d0b, ca_wv, ca_bv, nullptr, P, BNc, Dc, Dc, Dc);
    split_heads_kernel<<<splitBlocks, 256, 0, stream>>>(P, nullptr, F2, 1.f);
    gemm(d1b, ca_wv, ca_bv, nullptr, P, BNc, Dc, Dc, Dc);
    split_heads_kernel<<<splitBlocks, 256, 0, stream>>>(P, nullptr, F3, 1.f);

    attn_scores_kernel<<<dim3(Nc / 16, Nc / 256, BHc), 32, 0, stream>>>(F0, F1, attn01, 1.f);
    col_softmax_t_kernel<<<BHc * Nc, 256, 0, stream>>>(attn01, attn10);  // must precede in-place row softmax
    row_softmax_kernel<<<BHc * Nc, 256, 0, stream>>>(attn01);

    float* m0 = W2;
    float* m1 = W2 + (size_t)BNc * Dc;
    attn_av_kernel<<<dim3(Nc / 16, BHc), 32, 0, stream>>>(attn01, F3, m0);  // attn01 @ v1
    attn_av_kernel<<<dim3(Nc / 16, BHc), 32, 0, stream>>>(attn10, F2, m1);  // attn10 @ v0

    // x0 = d0 + ffn([d0, m0 @ wo + bo])
    gemm(m0, ca_wo, ca_bo, nullptr, W1 + Dc, BNc, Dc, Dc, 2 * Dc);
    copy_to_cat_kernel<<<(BNc * Dc) / 256, 256, 0, stream>>>(d0b, W1);
    gemm(W1, ca_w1, ca_b1, nullptr, W0, BNc, 2 * Dc, 2 * Dc, 2 * Dc);
    ln_gelu_kernel<<<BNc, 256, 0, stream>>>(W0, ca_g, ca_bn);
    gemm(W0, ca_w2, ca_b2, /*resid=*/d0b, x0_out, BNc, 2 * Dc, Dc, Dc);

    // x1 = d1 + ffn([d1, m1 @ wo + bo])
    gemm(m1, ca_wo, ca_bo, nullptr, W1 + Dc, BNc, Dc, Dc, 2 * Dc);
    copy_to_cat_kernel<<<(BNc * Dc) / 256, 256, 0, stream>>>(d1b, W1);
    gemm(W1, ca_w1, ca_b1, nullptr, W0, BNc, 2 * Dc, 2 * Dc, 2 * Dc);
    ln_gelu_kernel<<<BNc, 256, 0, stream>>>(W0, ca_g, ca_bn);
    gemm(W0, ca_w2, ca_b2, /*resid=*/d1b, x1_out, BNc, 2 * Dc, Dc, Dc);
}
